// SelfAttention_54485955117560
// MI455X (gfx1250) — matrix operand order repaired and checked by reading, not compiled, not run on hardware
//
#include <hip/hip_runtime.h>

// ---------------------------------------------------------------------------
// Self-attention forward for MI455X (gfx1250, wave32, WMMA bf16 path).
// B=4, S=2048, D=1024, heads=16, head_dim=64.
// ---------------------------------------------------------------------------

typedef __attribute__((ext_vector_type(16))) __bf16          v16bf;
typedef __attribute__((ext_vector_type(8)))  float           v8f;
typedef __attribute__((ext_vector_type(8)))  unsigned short  u16x8;
typedef __attribute__((ext_vector_type(16))) unsigned short  u16x16;

static constexpr int Dm   = 1024;   // model dim
static constexpr int Sq   = 2048;   // sequence length
static constexpr int Hh   = 16;     // heads
static constexpr int HD   = 64;     // head dim
static constexpr int Mrow = 4 * Sq; // 8192 token rows

__device__ __forceinline__ unsigned short f32_to_bf16(float f) {
  unsigned int u = __builtin_bit_cast(unsigned int, f);
  unsigned int r = u + 0x7FFFu + ((u >> 16) & 1u);  // round-to-nearest-even
  return (unsigned short)(r >> 16);
}

__device__ __forceinline__ u16x8 cvt8(float4 a, float4 b) {
  u16x8 h;
  h[0] = f32_to_bf16(a.x); h[1] = f32_to_bf16(a.y);
  h[2] = f32_to_bf16(a.z); h[3] = f32_to_bf16(a.w);
  h[4] = f32_to_bf16(b.x); h[5] = f32_to_bf16(b.y);
  h[6] = f32_to_bf16(b.z); h[7] = f32_to_bf16(b.w);
  return h;
}

__device__ __forceinline__ v16bf make_frag(u16x8 lo, u16x8 hi) {
  u16x16 v = __builtin_shufflevector(lo, hi, 0, 1, 2, 3, 4, 5, 6, 7,
                                             8, 9, 10, 11, 12, 13, 14, 15);
  return __builtin_bit_cast(v16bf, v);
}

// gfx1250 async global->LDS copy (16 bytes per lane), tracked by ASYNCcnt.
__device__ __forceinline__ unsigned lds_off32(const void* p) {
  return (unsigned)(uintptr_t)p;  // low 32 bits of an LDS pointer = LDS offset
}
__device__ __forceinline__ void async_copy_b128(unsigned lds_addr, const void* g) {
  asm volatile("global_load_async_to_lds_b128 %0, %1, off"
               :: "v"(lds_addr), "v"((unsigned long long)(uintptr_t)g)
               : "memory");
}
__device__ __forceinline__ void wait_async0() {
  asm volatile("s_wait_asynccnt 0x0" ::: "memory");
}

// ---------------------------------------------------------------------------
// Tiled GEMM: Out = act(A @ W^T + bias).   A:[8192,1024], W:[1024,1024] fp32.
// OUTMODE 0: bf16 out, layout [B,h,S,d]   (Q / K, scale folded in)
// OUTMODE 1: bf16 out, layout [B,h,d,S]   (V transposed for P*V B-fragments)
// OUTMODE 2: fp32 out, layout [8192,1024] (final projection)
// A_BF16: A is bf16 (ctx) instead of fp32 (x).
// Block tile 128(M) x 64(N), 8 waves, each wave 32x32 = 2x2 WMMA frags.
// Register-prefetch pipeline hides global latency behind WMMA.
// ---------------------------------------------------------------------------
template <int OUTMODE, bool A_BF16>
__global__ __launch_bounds__(256) void gemm_kernel(const void* __restrict__ Aptr,
                                                   const float* __restrict__ W,
                                                   const float* __restrict__ bias,
                                                   void* __restrict__ Out,
                                                   float scale) {
  constexpr int LDA = 48;  // padded ushort stride (96B: 16B-aligned rows)
  __shared__ unsigned short lds_a[128 * LDA];
  __shared__ unsigned short lds_b[64 * LDA];

  const int tid   = threadIdx.x;
  const int lane  = tid & 31;
  const int w     = tid >> 5;
  const int waveM = w & 3;   // 4 sub-tiles of 32 rows
  const int waveN = w >> 2;  // 2 sub-tiles of 32 cols
  const int half  = lane >> 4;
  const int l16   = lane & 15;
  const int bm    = blockIdx.y * 128;
  const int bn    = blockIdx.x * 64;

  // staging assignments: A has 512 16B tasks (2 per thread), B has 256.
  const int arow0 = tid >> 2,          acol0 = (tid & 3) * 8;
  const int arow1 = (tid + 256) >> 2,  acol1 = ((tid + 256) & 3) * 8;
  const int brow  = tid >> 2,          bcol  = (tid & 3) * 8;

  float4 arf[2][2], brf[2];  // fp32 A-path prefetch regs
  u16x8  arh[2];             // bf16 A-path prefetch regs

  auto load_regs = [&](int kc) {
    if constexpr (!A_BF16) {
      const float* A = (const float*)Aptr;
      const float* s0 = A + (size_t)(bm + arow0) * Dm + kc + acol0;
      const float* s1 = A + (size_t)(bm + arow1) * Dm + kc + acol1;
      arf[0][0] = *(const float4*)s0; arf[0][1] = *(const float4*)(s0 + 4);
      arf[1][0] = *(const float4*)s1; arf[1][1] = *(const float4*)(s1 + 4);
    } else {
      const unsigned short* A = (const unsigned short*)Aptr;
      arh[0] = *(const u16x8*)(A + (size_t)(bm + arow0) * Dm + kc + acol0);
      arh[1] = *(const u16x8*)(A + (size_t)(bm + arow1) * Dm + kc + acol1);
    }
    const float* sb = W + (size_t)(bn + brow) * Dm + kc + bcol;
    brf[0] = *(const float4*)sb; brf[1] = *(const float4*)(sb + 4);
  };

  v8f acc[2][2] = {};
  load_regs(0);  // prologue prefetch

  for (int kc = 0; kc < Dm; kc += 32) {
    __syncthreads();  // previous iteration's LDS reads complete
    if constexpr (!A_BF16) {
      *(u16x8*)&lds_a[arow0 * LDA + acol0] = cvt8(arf[0][0], arf[0][1]);
      *(u16x8*)&lds_a[arow1 * LDA + acol1] = cvt8(arf[1][0], arf[1][1]);
    } else {
      *(u16x8*)&lds_a[arow0 * LDA + acol0] = arh[0];
      *(u16x8*)&lds_a[arow1 * LDA + acol1] = arh[1];
    }
    *(u16x8*)&lds_b[brow * LDA + bcol] = cvt8(brf[0], brf[1]);
    __syncthreads();

    if (kc + 32 < Dm) load_regs(kc + 32);  // overlap next slab with WMMA below

    // ---- fragments + WMMA ----
    v16bf afrag[2], bfrag[2];
#pragma unroll
    for (int fm = 0; fm < 2; ++fm) {
      int r = (waveM * 32 + fm * 16 + l16) * LDA;
      u16x8 lo = *(const u16x8*)&lds_a[r + half * 8];        // K 0..7 / 8..15
      u16x8 hi = *(const u16x8*)&lds_a[r + 16 + half * 8];   // K 16..23 / 24..31
      afrag[fm] = make_frag(lo, hi);
    }
#pragma unroll
    for (int fn = 0; fn < 2; ++fn) {
      int c = (waveN * 32 + fn * 16 + l16) * LDA;
      u16x8 lo = *(const u16x8*)&lds_b[c + half * 8];
      u16x8 hi = *(const u16x8*)&lds_b[c + 16 + half * 8];
      bfrag[fn] = make_frag(lo, hi);
    }
#pragma unroll
    for (int fm = 0; fm < 2; ++fm)
#pragma unroll
      for (int fn = 0; fn < 2; ++fn)
        acc[fm][fn] = __builtin_amdgcn_wmma_f32_16x16x32_bf16(
            false, afrag[fm], false, bfrag[fn], (short)0, acc[fm][fn], false, false);
  }

  // ---- epilogue ----
#pragma unroll
  for (int fm = 0; fm < 2; ++fm) {
#pragma unroll
    for (int fn = 0; fn < 2; ++fn) {
      int row0 = bm + waveM * 32 + fm * 16 + half * 8;
      int col  = bn + waveN * 32 + fn * 16 + l16;
      float bv = bias[col];
      if constexpr (OUTMODE == 0) {  // Q/K: bf16 [B,h,S,d]
        unsigned short* O = (unsigned short*)Out;
        int head = col >> 6, dd = col & 63;
#pragma unroll
        for (int i = 0; i < 8; ++i) {
          int row = row0 + i, b = row >> 11, s = row & 2047;
          O[(((size_t)(b * Hh + head) * Sq + s) << 6) + dd] =
              f32_to_bf16((acc[fm][fn][i] + bv) * scale);
        }
      } else if constexpr (OUTMODE == 1) {  // V: bf16 [B,h,d,S] (transposed)
        unsigned short* O = (unsigned short*)Out;
        int head = col >> 6, dd = col & 63;
        int b = row0 >> 11, s0 = row0 & 2047;
        u16x8 pack;
#pragma unroll
        for (int i = 0; i < 8; ++i) pack[i] = f32_to_bf16(acc[fm][fn][i] + bv);
        *(u16x8*)&O[((size_t)((b * Hh + head) * HD + dd) << 11) + s0] = pack;
      } else {  // fp32 [8192,1024]
        float* O = (float*)Out;
#pragma unroll
        for (int i = 0; i < 8; ++i)
          O[(size_t)(row0 + i) * Dm + col] = acc[fm][fn][i] + bv;
      }
    }
  }
}

// ---------------------------------------------------------------------------
// Flash attention: ctx[b,s,h*d] = softmax(Q K^T + mask) V   (scale folded into
// Q and K already).  Grid: (S/128, B*h).  8 waves x 16 query rows each.
// K/V/mask tiles double-buffered in LDS via async global->LDS copies; chunk
// i+1's DMA overlaps chunk i's WMMA + softmax.
// ---------------------------------------------------------------------------
__global__ __launch_bounds__(256) void attn_kernel(const unsigned short* __restrict__ Q,
                                                   const unsigned short* __restrict__ K,
                                                   const unsigned short* __restrict__ Vt,
                                                   const unsigned char* __restrict__ mask,
                                                   unsigned short* __restrict__ Ctx) {
  constexpr int LDP = 48;  // padded P-scratch stride (ushorts)
  constexpr int LDK = 72;  // K tile stride: 144B, 16B-aligned, conflict-free
  constexpr int LDV = 40;  // V tile stride: 80B,  16B-aligned, conflict-free
  constexpr int KT = 32 * LDK, VT = 64 * LDV, MT = 128 * 32;
  __shared__ unsigned short k_tile[2 * KT];       // [buf][key][d]
  __shared__ unsigned short v_tile[2 * VT];       // [buf][d][s]
  __shared__ unsigned char  m_tile[2 * MT];       // [buf][qrow][key]
  __shared__ unsigned short lds_p[8 * 16 * LDP];  // per-wave 16x32 P tile

  const int tid  = threadIdx.x;
  const int lane = tid & 31;
  const int w    = tid >> 5;
  const int half = lane >> 4;
  const int l16  = lane & 15;
  const int bh   = blockIdx.y;
  const int b    = bh >> 4;
  const int head = bh & 15;
  const int q0   = blockIdx.x * 128 + w * 16;  // this wave's 16 query rows

  const unsigned short* Qb = Q + (size_t)bh * Sq * HD;   // [s][d]
  const unsigned short* Kb = K + (size_t)bh * Sq * HD;   // [s][d]
  const unsigned short* Vb = Vt + (size_t)bh * HD * Sq;  // [d][s]
  const unsigned char*  mb = mask + (size_t)b * Sq * Sq;
  unsigned short* pw = &lds_p[w * 16 * LDP];

  // Q A-fragments (16x64 -> two 16x32), kept resident.
  v16bf qfrag[2];
#pragma unroll
  for (int c = 0; c < 2; ++c) {
    const unsigned short* rowp = Qb + (size_t)(q0 + l16) * HD + c * 32;
    u16x8 lo = *(const u16x8*)(rowp + half * 8);
    u16x8 hi = *(const u16x8*)(rowp + 16 + half * 8);
    qfrag[c] = make_frag(lo, hi);
  }

  // per-thread staging assignments (16B per thread per tile)
  const int krow = tid >> 3, kseg = tid & 7;    // K: 32 rows x 8 segs
  const int vrow = tid >> 2, vseg = tid & 3;    // V: 64 rows x 4 segs
  const int mrw  = tid >> 1, mseg = tid & 1;    // mask: 128 rows x 2 segs

  auto issue_tile = [&](int k0, int buf) {
    async_copy_b128(lds_off32(&k_tile[buf * KT + krow * LDK + kseg * 8]),
                    Kb + (size_t)(k0 + krow) * HD + kseg * 8);
    async_copy_b128(lds_off32(&v_tile[buf * VT + vrow * LDV + vseg * 8]),
                    Vb + (size_t)vrow * Sq + k0 + vseg * 8);
    async_copy_b128(lds_off32(&m_tile[buf * MT + mrw * 32 + mseg * 16]),
                    mb + (size_t)(blockIdx.x * 128 + mrw) * Sq + k0 + mseg * 16);
  };

  float mrow[8], lrow[8];
#pragma unroll
  for (int i = 0; i < 8; ++i) { mrow[i] = -INFINITY; lrow[i] = 0.f; }
  v8f acc[4] = {};

  issue_tile(0, 0);  // prologue DMA
  int buf = 0;

  for (int k0 = 0; k0 < Sq; k0 += 32) {
    wait_async0();    // current tile landed (only its 3 copies outstanding)
    __syncthreads();  // ...for every wave in the workgroup
    if (k0 + 32 < Sq) issue_tile(k0 + 32, buf ^ 1);  // DMA under compute

    const unsigned short* kt = &k_tile[buf * KT];
    const unsigned short* vt = &v_tile[buf * VT];
    const unsigned char*  mt = &m_tile[buf * MT];

    // ---- scores: two 16x16 tiles (keys k0..k0+31) ----
    v8f sc[2] = {};
#pragma unroll
    for (int h2 = 0; h2 < 2; ++h2) {
      const unsigned short* kp = kt + (h2 * 16 + l16) * LDK;
#pragma unroll
      for (int c = 0; c < 2; ++c) {
        u16x8 lo = *(const u16x8*)(kp + c * 32 + half * 8);
        u16x8 hi = *(const u16x8*)(kp + c * 32 + 16 + half * 8);
        v16bf kf = make_frag(lo, hi);
        sc[h2] = __builtin_amdgcn_wmma_f32_16x16x32_bf16(
            false, qfrag[c], false, kf, (short)0, sc[h2], false, false);
      }
    }
    // ---- mask (from LDS tile) ----
#pragma unroll
    for (int h2 = 0; h2 < 2; ++h2)
#pragma unroll
      for (int i = 0; i < 8; ++i) {
        int rloc = w * 16 + half * 8 + i;
        if (!mt[rloc * 32 + h2 * 16 + l16]) sc[h2][i] = -1e9f;
      }
    // ---- online softmax (rows live across 16 lanes of each half) ----
    float corr[8];
#pragma unroll
    for (int i = 0; i < 8; ++i) {
      float t = fmaxf(sc[0][i], sc[1][i]);
#pragma unroll
      for (int off = 1; off < 16; off <<= 1) t = fmaxf(t, __shfl_xor(t, off, 32));
      float mnew = fmaxf(mrow[i], t);
      corr[i] = __expf(mrow[i] - mnew);
      mrow[i] = mnew;
      float p0 = __expf(sc[0][i] - mnew);
      float p1 = __expf(sc[1][i] - mnew);
      sc[0][i] = p0; sc[1][i] = p1;
      float rs = p0 + p1;
#pragma unroll
      for (int off = 1; off < 16; off <<= 1) rs += __shfl_xor(rs, off, 32);
      lrow[i] = lrow[i] * corr[i] + rs;
    }
#pragma unroll
    for (int n = 0; n < 4; ++n)
#pragma unroll
      for (int i = 0; i < 8; ++i) acc[n][i] *= corr[i];

    // ---- relayout P: C-fragment -> A-fragment via per-wave LDS ----
#pragma unroll
    for (int i = 0; i < 8; ++i) {
      int row = half * 8 + i;
      pw[row * LDP + l16]      = f32_to_bf16(sc[0][i]);
      pw[row * LDP + 16 + l16] = f32_to_bf16(sc[1][i]);
    }
    __builtin_amdgcn_wave_barrier();
    u16x8 plo = *(const u16x8*)&pw[l16 * LDP + half * 8];
    u16x8 phi = *(const u16x8*)&pw[l16 * LDP + 16 + half * 8];
    v16bf pfrag = make_frag(plo, phi);

    // ---- out += P(16x32) * V(32x64), four 16-wide d tiles ----
#pragma unroll
    for (int n = 0; n < 4; ++n) {
      const unsigned short* vp = vt + (n * 16 + l16) * LDV + half * 8;
      u16x8 vlo = *(const u16x8*)(vp);
      u16x8 vhi = *(const u16x8*)(vp + 16);
      v16bf vf = make_frag(vlo, vhi);
      acc[n] = __builtin_amdgcn_wmma_f32_16x16x32_bf16(
          false, pfrag, false, vf, (short)0, acc[n], false, false);
    }
    __syncthreads();  // all waves done reading buf before it is re-DMA'd
    buf ^= 1;
  }

  // ---- normalize and write ctx (bf16 [B,S,h*d]) ----
#pragma unroll
  for (int n = 0; n < 4; ++n) {
    int dd = head * HD + n * 16 + l16;
#pragma unroll
    for (int i = 0; i < 8; ++i) {
      int row = q0 + half * 8 + i;
      float o = acc[n][i] / lrow[i];
      Ctx[((size_t)b * Sq + row) * Dm + dd] = f32_to_bf16(o);
    }
  }
}

// ---------------------------------------------------------------------------
extern "C" void kernel_launch(void* const* d_in, const int* in_sizes, int n_in,
                              void* d_out, int out_size, void* d_ws, size_t ws_size,
                              hipStream_t stream) {
  const float*         x    = (const float*)d_in[0];
  const unsigned char* mask = (const unsigned char*)d_in[1];
  const float* Wq = (const float*)d_in[2];
  const float* bq = (const float*)d_in[3];
  const float* Wk = (const float*)d_in[4];
  const float* bk = (const float*)d_in[5];
  const float* Wv = (const float*)d_in[6];
  const float* bv = (const float*)d_in[7];
  const float* Wo = (const float*)d_in[8];
  const float* bo = (const float*)d_in[9];

  // workspace: Q, K, Vt, Ctx each 8192*1024 bf16 = 16 MiB (64 MiB total)
  const size_t N = (size_t)Mrow * Dm;
  unsigned short* qws = (unsigned short*)d_ws;
  unsigned short* kws = qws + N;
  unsigned short* vws = kws + N;
  unsigned short* cws = vws + N;

  const float scale = 0.35355339059327379f;  // 64^-0.25, applied to Q and K
  dim3 ggrid(Dm / 64, Mrow / 128);           // (16, 64)
  dim3 agrid(Sq / 128, 4 * Hh);              // (16, 64)

  gemm_kernel<0, false><<<ggrid, 256, 0, stream>>>(x, Wq, bq, qws, scale);
  gemm_kernel<0, false><<<ggrid, 256, 0, stream>>>(x, Wk, bk, kws, scale);
  gemm_kernel<1, false><<<ggrid, 256, 0, stream>>>(x, Wv, bv, vws, 1.0f);
  attn_kernel<<<agrid, 256, 0, stream>>>(qws, kws, vws, mask, cws);
  gemm_kernel<2, true><<<ggrid, 256, 0, stream>>>(cws, Wo, bo, d_out, 1.0f);
}
